// GPTLanguageModel_30279519437191
// MI455X (gfx1250) — compile-verified, matrix operand order checked by reading, not run on hardware
//
#include <hip/hip_runtime.h>

// ---------------------------------------------------------------------------
// GPT forward for MI455X (gfx1250): bf16 WMMA (v_wmma_f32_16x16x32_bf16) with
// f32 accumulate; GEMM operand tiles streamed into LDS by the Tensor Data
// Mover (tensor_load_to_lds, TENSORcnt, double-buffered) so DMA overlaps the
// matrix pipe. Compute-bound (~0.78 TFLOP); bf16 weights (~21MB) stay in L2.
// ---------------------------------------------------------------------------

typedef unsigned short u16;
typedef __attribute__((ext_vector_type(16))) __bf16 v16bf;
typedef __attribute__((ext_vector_type(8)))  float  v8f;
typedef __attribute__((ext_vector_type(4)))  unsigned int u32x4;
typedef __attribute__((ext_vector_type(8)))  int i32x8;
typedef __attribute__((ext_vector_type(4)))  int i32x4;

union FragB16 { v16bf v; uint4 q[2]; u16 us[16]; };

#if defined(__gfx1250__) && __has_builtin(__builtin_amdgcn_tensor_load_to_lds) && __has_builtin(__builtin_amdgcn_s_wait_tensorcnt)
#define HAVE_TDM 1
#else
#define HAVE_TDM 0
#endif

__device__ __forceinline__ u16 f2bf(float f) {
  union { float f; unsigned u; } x; x.f = f;
  unsigned u = x.u;
  return (u16)((u + 0x7FFFu + ((u >> 16) & 1u)) >> 16);   // round-nearest-even
}

static constexpr int Bsz = 128, Tsz = 256, Esz = 384, Hsz = 6, HSsz = 64,
                     Lsz = 6, Vsz = 78, FFsz = 1536;
static constexpr int Msz = Bsz * Tsz;           // 32768 tokens
static constexpr int QKVW = Hsz * 3 * HSsz;     // 1152

#if HAVE_TDM
// TDM 2D tile load: global (row-major, row_stride elems of 2B) -> LDS with
// 16B padding appended after every 64B row (LDS row stride = 40 bf16 elems).
// D# per ISA 8.3-8.6: group0 = {ctrl, lds_addr, gaddr lo, gaddr hi | type=2},
// group1 packs data_size=2B, pad_interval=16DW, pad_amount=4DW, tile dims,
// tensor dims (huge -> no OOB clamp), dim0 stride. Groups 2/3 unused (2D).
__device__ __forceinline__ void tdm_load_2d(unsigned lds_byte_off,
                                            const u16* gaddr,
                                            unsigned tile_k,     // dim0 elems
                                            unsigned tile_rows,  // dim1
                                            unsigned row_stride) {
  unsigned long long ga = (unsigned long long)gaddr;
  u32x4 g0;
  g0.x = 1u;                                            // count=1 (valid user D#)
  g0.y = lds_byte_off;                                  // LDS dest (bytes)
  g0.z = (unsigned)(ga & 0xFFFFFFFFu);                  // global_addr[31:0]
  g0.w = (unsigned)((ga >> 32) & 0x01FFFFFFu) | (2u << 30);  // [56:32] | type=2
  i32x8 g1;
  g1[0] = (int)((1u << 16) | (1u << 20) | (3u << 22) | (3u << 25));
  //            data_size=2B  pad_en      pad_int=16DW  pad_amt=4DW
  g1[1] = (int)(0xFFFFu << 16);                         // tensor_dim0 lo16
  g1[2] = (int)(0x7FFFu | (0xFFFFu << 16));             // dim0 hi16 | dim1 lo16
  g1[3] = (int)(0x7FFFu | ((tile_k & 0xFFFFu) << 16));  // dim1 hi16 | tile_dim0
  g1[4] = (int)(tile_rows & 0xFFFFu);                   // tile_dim1 | tile_dim2=0
  g1[5] = (int)row_stride;                              // tensor_dim0_stride lo32
  g1[6] = 0;                                            // stride hi | dim1_stride
  g1[7] = 0;
  i32x4 z4 = {0, 0, 0, 0};
  i32x8 z8 = {0, 0, 0, 0, 0, 0, 0, 0};
  __builtin_amdgcn_tensor_load_to_lds(g0, g1, z4, z4, z8, 0);
}
#endif

// ---------------------------------------------------------------------------
// Embedding: x[m,e] = tok_emb[ctx[m], e] + pos_emb[m % T, e]
// ---------------------------------------------------------------------------
__global__ __launch_bounds__(256)
void k_embed(const int* __restrict__ ctx, const float* __restrict__ tok,
             const float* __restrict__ pos, float* __restrict__ x) {
  long i = (long)blockIdx.x * 256 + threadIdx.x;
  if (i >= (long)Msz * Esz) return;
  int m = (int)(i / Esz), e = (int)(i - (long)m * Esz);
  int t = m % Tsz;
  x[i] = tok[(long)ctx[m] * Esz + e] + pos[(long)t * Esz + e];
}

// ---------------------------------------------------------------------------
// Weight converters: fp32 [.., K, N] -> bf16 TRANSPOSED [.., N, K]
// ---------------------------------------------------------------------------
__global__ __launch_bounds__(256)
void k_cvt_t(const float* __restrict__ src, u16* __restrict__ dst,
             int nl, int K, int N) {
  long i = (long)blockIdx.x * 256 + threadIdx.x;
  if (i >= (long)nl * N * K) return;
  int l = (int)(i / ((long)N * K));
  long r = i - (long)l * N * K;
  int n = (int)(r / K);
  int k = (int)(r - (long)n * K);
  dst[i] = f2bf(src[((long)l * K + k) * N + n]);
}

// w_qkv [L,H,E,192] -> per-layer transposed [H*192, E]
__global__ __launch_bounds__(256)
void k_cvt_qkv_t(const float* __restrict__ src, u16* __restrict__ dst) {
  long i = (long)blockIdx.x * 256 + threadIdx.x;
  if (i >= (long)Lsz * QKVW * Esz) return;
  int l = (int)(i / ((long)QKVW * Esz));
  long r = i - (long)l * QKVW * Esz;
  int n = (int)(r / Esz);                 // h*192 + f
  int k = (int)(r - (long)n * Esz);       // e
  int h = n / 192, f = n - h * 192;
  dst[i] = f2bf(src[(((long)l * Hsz + h) * Esz + k) * 192 + f]);
}

// w_head [E,78] -> transposed+padded [128, E]
__global__ __launch_bounds__(256)
void k_cvt_head_t(const float* __restrict__ src, u16* __restrict__ dst) {
  long i = (long)blockIdx.x * 256 + threadIdx.x;
  if (i >= (long)128 * Esz) return;
  int n = (int)(i / Esz), k = (int)(i - (long)n * Esz);
  dst[i] = (n < Vsz) ? f2bf(src[(long)k * Vsz + n]) : (u16)0;
}

// V transpose: qkv [M, 1152] -> vT [B,H,HS,T] (key index contiguous)
__global__ __launch_bounds__(256)
void k_vtrans(const u16* __restrict__ qkv, u16* __restrict__ vT) {
  long i = (long)blockIdx.x * 256 + threadIdx.x;
  if (i >= (long)Bsz * Hsz * HSsz * Tsz) return;
  int t  = (int)(i & (Tsz - 1));
  long r = i >> 8;
  int hs = (int)(r & (HSsz - 1));
  long r2 = r >> 6;
  int h = (int)(r2 % Hsz);
  int b = (int)(r2 / Hsz);
  vT[i] = qkv[((long)b * Tsz + t) * QKVW + h * 192 + 128 + hs];
}

// ---------------------------------------------------------------------------
// LayerNorm -> bf16, one wave32 per row (E=384 -> 12 elems/lane)
// ---------------------------------------------------------------------------
__global__ __launch_bounds__(128)
void k_ln_bf16(const float* __restrict__ x, const float* __restrict__ g,
               const float* __restrict__ b, u16* __restrict__ out, int ld_out) {
  int row  = blockIdx.x * 4 + (threadIdx.x >> 5);
  int lane = threadIdx.x & 31;
  const float* xr = x + (long)row * Esz;
  float v[12], s = 0.f;
#pragma unroll
  for (int j = 0; j < 12; ++j) { v[j] = xr[lane + 32 * j]; s += v[j]; }
#pragma unroll
  for (int off = 16; off > 0; off >>= 1) s += __shfl_xor(s, off, 32);
  float mu = s * (1.0f / Esz), q = 0.f;
#pragma unroll
  for (int j = 0; j < 12; ++j) { float d = v[j] - mu; q += d * d; }
#pragma unroll
  for (int off = 16; off > 0; off >>= 1) q += __shfl_xor(q, off, 32);
  float rs = rsqrtf(q * (1.0f / Esz) + 1e-5f);
  u16* o = out + (long)row * ld_out;
#pragma unroll
  for (int j = 0; j < 12; ++j) {
    int e = lane + 32 * j;
    o[e] = f2bf((v[j] - mu) * rs * g[e] + b[e]);
  }
}

// ---------------------------------------------------------------------------
// bf16 WMMA GEMM: C[M,N] = A[M,K] * Bt[N,K]^T (+bias)(+resid)(ReLU?)
// block = 256 threads (8 waves), tile 128x64, K-step 32.
// TDM path: wave 0 streams the next K-step's A/B tiles into the alternate
// LDS buffer (tensor_load_to_lds) while all 8 waves run WMMA on the current
// one; s_wait_tensorcnt(2) retires the current pair (per-wave in-order TDM).
// Fallback path: vectorized manual staging (b128 loads/stores) + prefetch.
// ---------------------------------------------------------------------------
#define LDK_S 40   // 32 K + 8 pad (bf16); row = 64B data + 16B pad, 16B-aligned
__global__ __launch_bounds__(256)
void k_gemm_bf16(const u16* __restrict__ A, int lda,
                 const u16* __restrict__ Bt, int ldb,   // ldb == K
                 const float* __restrict__ bias,
                 const float* __restrict__ resid, int ldr,
                 float* __restrict__ Cf, int ldc,
                 u16* __restrict__ Cb, int ldcb,
                 int K, int Nstore, int relu) {
  __shared__ __align__(16) u16 As[2][128 * LDK_S];
  __shared__ __align__(16) u16 Bs[2][64 * LDK_S];
  int tid = threadIdx.x;
  int wave = tid >> 5, lane = tid & 31;
  int r15 = lane & 15, hi = lane >> 4;
  int rowBlock = blockIdx.y * 128;
  int colBlock = blockIdx.x * 64;

  const u16* Ablk = A + (long)rowBlock * lda;
  const u16* Bblk = Bt + (long)colBlock * ldb;
  int nsteps = K >> 5;

  v8f acc[4] = {};

#if HAVE_TDM
  if (wave == 0) {
    tdm_load_2d((unsigned)(unsigned long long)(void*)&As[0][0], Ablk, 32, 128, (unsigned)lda);
    tdm_load_2d((unsigned)(unsigned long long)(void*)&Bs[0][0], Bblk, 32, 64, (unsigned)ldb);
  }
#else
  int ar = tid >> 1;            // 0..127 A rows, 16 elems each (2x uint4)
  int ac = (tid & 1) * 16;
  int nr = tid >> 2;            // 0..63  B rows (N), 8 elems each (1x uint4)
  int nc = (tid & 3) * 8;
#endif

  for (int i = 0; i < nsteps; ++i) {
    int buf = i & 1;
#if HAVE_TDM
    if (wave == 0) {
      if (i + 1 < nsteps) {
        tdm_load_2d((unsigned)(unsigned long long)(void*)&As[buf ^ 1][0],
                    Ablk + (i + 1) * 32, 32, 128, (unsigned)lda);
        tdm_load_2d((unsigned)(unsigned long long)(void*)&Bs[buf ^ 1][0],
                    Bblk + (i + 1) * 32, 32, 64, (unsigned)ldb);
        __builtin_amdgcn_s_wait_tensorcnt(2);   // pair i done; pair i+1 in flight
      } else {
        __builtin_amdgcn_s_wait_tensorcnt(0);   // last pair done
      }
    }
#else
    {
      int k0 = i * 32;
      const u16* aptr = Ablk + (long)ar * lda + k0 + ac;
      *(uint4*)&As[buf][ar * LDK_S + ac]     = *(const uint4*)aptr;
      *(uint4*)&As[buf][ar * LDK_S + ac + 8] = *(const uint4*)(aptr + 8);
      const u16* bptr = Bblk + (long)nr * ldb + k0 + nc;
      *(uint4*)&Bs[buf][nr * LDK_S + nc] = *(const uint4*)bptr;
      if (i + 1 < nsteps) {
        __builtin_prefetch(aptr + 32, 0, 0);
        __builtin_prefetch(bptr + 32, 0, 0);
      }
    }
#endif
    __syncthreads();

    // A fragment: lane[3:0]=row, lane[4]=K-half; two contiguous 8-elem chunks
    FragB16 fa;
    {
      const u16* base = &As[buf][(wave * 16 + r15) * LDK_S + hi * 8];
      fa.q[0] = *(const uint4*)base;
      fa.q[1] = *(const uint4*)(base + 16);
    }
#pragma unroll
    for (int nt = 0; nt < 4; ++nt) {
      FragB16 fb;                // lane[3:0]=col, same K-chunk pattern
      const u16* bbase = &Bs[buf][(nt * 16 + r15) * LDK_S + hi * 8];
      fb.q[0] = *(const uint4*)bbase;
      fb.q[1] = *(const uint4*)(bbase + 16);
      acc[nt] = __builtin_amdgcn_wmma_f32_16x16x32_bf16(
          false, fa.v, false, fb.v, (short)0, acc[nt], false, false);
    }
    __syncthreads();   // all waves done with buf before DMA/staging reuses it
  }

  // epilogue: C layout VGPR r -> row = 8*hi + r, col = 16*nt + r15
#pragma unroll
  for (int nt = 0; nt < 4; ++nt) {
    int col = colBlock + nt * 16 + r15;
    float bia = (bias != nullptr && col < Nstore) ? bias[col] : 0.f;
#pragma unroll
    for (int r = 0; r < 8; ++r) {
      int row = rowBlock + wave * 16 + hi * 8 + r;
      float val = acc[nt][r] + bia;
      if (resid) val += resid[(long)row * ldr + col];
      if (relu)  val = fmaxf(val, 0.f);
      if (Cf && col < Nstore) Cf[(long)row * ldc + col] = val;
      if (Cb) Cb[(long)row * ldcb + col] = f2bf(val);
    }
  }
}

// ---------------------------------------------------------------------------
// Attention: one wave per (b, h, 16 query rows). Causal => only key tiles
// <= query tile. Q/K^T fragments are contiguous loads from the qkv buffer;
// V fragments are contiguous loads from the pre-transposed vT buffer.
// qkv per-token layout: [h][q 0:64 | k 64:128 | v 128:192]
// ---------------------------------------------------------------------------
#define SROW 260
#define PROW 264
__global__ __launch_bounds__(32)
void k_attn(const u16* __restrict__ qkv, const u16* __restrict__ vT,
            u16* __restrict__ outA) {
  __shared__ __align__(16) float S[16 * SROW];
  __shared__ __align__(16) u16   P[16 * PROW];

  int blk = blockIdx.x;
  int qt  = blk & 15;            // query tile (T/16 = 16)
  int bh  = blk >> 4;
  int h   = bh % Hsz;
  int b   = bh / Hsz;
  int lane = threadIdx.x;
  int r15 = lane & 15, hi = lane >> 4;

  const long tokS = QKVW;        // 1152
  const u16* qbase = qkv + ((long)b * Tsz) * tokS + h * 192;
  const u16* kbase = qbase + 64;
  const u16* vtb   = vT + ((long)(b * Hsz + h) * HSsz) * Tsz;
  int qrow0 = qt * 16;

  // Q fragments for both K-steps over HS=64
  FragB16 fq[2];
  {
    const u16* qp = qbase + (long)(qrow0 + r15) * tokS + hi * 8;
    fq[0].q[0] = *(const uint4*)qp;
    fq[0].q[1] = *(const uint4*)(qp + 16);
    fq[1].q[0] = *(const uint4*)(qp + 32);
    fq[1].q[1] = *(const uint4*)(qp + 48);
  }

  const float scale = 0.125f;    // 1/sqrt(64)
  int ktiles = qt + 1;
  for (int kt = 0; kt < ktiles; ++kt) {
    v8f c = {};
#pragma unroll
    for (int s = 0; s < 2; ++s) {
      FragB16 fk;                // B[k=hs][n=key] = K[key][hs], hs contiguous
      const u16* kp = kbase + (long)(kt * 16 + r15) * tokS + s * 32 + hi * 8;
      fk.q[0] = *(const uint4*)kp;
      fk.q[1] = *(const uint4*)(kp + 16);
      c = __builtin_amdgcn_wmma_f32_16x16x32_bf16(
          false, fq[s].v, false, fk.v, (short)0, c, false, false);
    }
#pragma unroll
    for (int r = 0; r < 8; ++r) {
      int qr  = hi * 8 + r;
      int key = kt * 16 + r15;
      float sv = c[r] * scale;
      if (key > qrow0 + qr) sv = -__builtin_inff();
      S[qr * SROW + key] = sv;
    }
  }
  __syncthreads();

  int kend = ktiles * 16;
  // softmax: lane handles row r15, key segment [hi*128, ...)
  int row = r15;
  int klo = hi * 128, khi = (hi + 1) * 128;
  if (khi > kend) khi = kend;
  float mx = -__builtin_inff();
  for (int k = klo; k < khi; ++k) mx = fmaxf(mx, S[row * SROW + k]);
  mx = fmaxf(mx, __shfl_xor(mx, 16, 32));
  float sum = 0.f;
  for (int k = klo; k < khi; ++k) {
    float p = __expf(S[row * SROW + k] - mx);
    sum += p;
    S[row * SROW + k] = p;
  }
  sum += __shfl_xor(sum, 16, 32);
  float inv = 1.0f / sum;
  for (int k = klo; k < khi; ++k)
    P[row * PROW + k] = f2bf(S[row * SROW + k] * inv);
  if (kend < Tsz) {              // zero tail half-block for the P@V loop
#pragma unroll
    for (int j = 0; j < 8; ++j) P[r15 * PROW + kend + hi * 8 + j] = 0;
  }
  __syncthreads();

  // O = P @ V, key blocks of 32; V fragments direct from vT (contiguous keys)
  v8f o[4] = {};
  int kblocks = (kend + 31) >> 5;
  for (int kb = 0; kb < kblocks; ++kb) {
    FragB16 fp;
    {
      const u16* pp = &P[r15 * PROW + kb * 32 + hi * 8];
      fp.q[0] = *(const uint4*)pp;
      fp.q[1] = *(const uint4*)(pp + 16);
    }
#pragma unroll
    for (int nt = 0; nt < 4; ++nt) {
      FragB16 fv;                // B[k=key][n=hs] = vT[hs][key], key contiguous
      const u16* vp = vtb + (long)(nt * 16 + r15) * Tsz + kb * 32 + hi * 8;
      fv.q[0] = *(const uint4*)vp;
      fv.q[1] = *(const uint4*)(vp + 16);
      o[nt] = __builtin_amdgcn_wmma_f32_16x16x32_bf16(
          false, fp.v, false, fv.v, (short)0, o[nt], false, false);
    }
  }

  // store O (bf16) into [M, 384] activation buffer at col h*64 + hs
#pragma unroll
  for (int nt = 0; nt < 4; ++nt)
#pragma unroll
    for (int r = 0; r < 8; ++r) {
      int qr = qrow0 + hi * 8 + r;
      int hs = nt * 16 + r15;
      outA[((long)b * Tsz + qr) * Esz + h * HSsz + hs] = f2bf(o[nt][r]);
    }
}

// ---------------------------------------------------------------------------
// Host orchestration
// ---------------------------------------------------------------------------
extern "C" void kernel_launch(void* const* d_in, const int* in_sizes, int n_in,
                              void* d_out, int out_size, void* d_ws, size_t ws_size,
                              hipStream_t stream) {
  const int*   ctx    = (const int*)  d_in[0];
  const float* tok    = (const float*)d_in[1];
  const float* pos    = (const float*)d_in[2];
  const float* ln1_g  = (const float*)d_in[3];
  const float* ln1_b  = (const float*)d_in[4];
  const float* w_qkv  = (const float*)d_in[5];
  const float* w_proj = (const float*)d_in[6];
  const float* b_proj = (const float*)d_in[7];
  const float* ln2_g  = (const float*)d_in[8];
  const float* ln2_b  = (const float*)d_in[9];
  const float* w1     = (const float*)d_in[10];
  const float* b1     = (const float*)d_in[11];
  const float* w2     = (const float*)d_in[12];
  const float* b2     = (const float*)d_in[13];
  const float* lnf_g  = (const float*)d_in[14];
  const float* lnf_b  = (const float*)d_in[15];
  const float* w_head = (const float*)d_in[16];
  const float* b_head = (const float*)d_in[17];

  char* w = (char*)d_ws;
  auto carve = [&](size_t bytes) -> void* {
    void* p = (void*)w; w += (bytes + 255) & ~(size_t)255; return p;
  };
  float* x     = (float*)carve((size_t)Msz * Esz * 4);        // residual stream
  u16* bufA    = (u16*)carve((size_t)Msz * Esz * 2);          // ln outs / attn O
  u16* bufB    = (u16*)carve((size_t)Msz * FFsz * 2);         // qkv / ff hidden
  u16* vTbuf   = (u16*)carve((size_t)Bsz * Hsz * HSsz * Tsz * 2);
  u16* wqkv_t  = (u16*)carve((size_t)Lsz * QKVW * Esz * 2);
  u16* wproj_t = (u16*)carve((size_t)Lsz * Esz * Esz * 2);
  u16* w1_t    = (u16*)carve((size_t)Lsz * FFsz * Esz * 2);
  u16* w2_t    = (u16*)carve((size_t)Lsz * Esz * FFsz * 2);
  u16* whead_t = (u16*)carve((size_t)128 * Esz * 2);
  (void)ws_size; (void)n_in; (void)in_sizes; (void)out_size;

  auto blocks = [](long n) { return (unsigned)((n + 255) / 256); };

  // embedding + weight conversion (all weights -> bf16 transposed [N][K])
  k_embed<<<blocks((long)Msz * Esz), 256, 0, stream>>>(ctx, tok, pos, x);
  k_cvt_qkv_t<<<blocks((long)Lsz * QKVW * Esz), 256, 0, stream>>>(w_qkv, wqkv_t);
  k_cvt_t<<<blocks((long)Lsz * Esz * Esz), 256, 0, stream>>>(w_proj, wproj_t, Lsz, Esz, Esz);
  k_cvt_t<<<blocks((long)Lsz * FFsz * Esz), 256, 0, stream>>>(w1, w1_t, Lsz, Esz, FFsz);
  k_cvt_t<<<blocks((long)Lsz * Esz * FFsz), 256, 0, stream>>>(w2, w2_t, Lsz, FFsz, Esz);
  k_cvt_head_t<<<blocks((long)128 * Esz), 256, 0, stream>>>(w_head, whead_t);

  auto gemm = [&](const u16* A, int lda, const u16* Bt, int N, int K,
                  const float* bias, const float* resid, int ldr,
                  float* Cf, int ldc, u16* Cb, int ldcb, int Nstore, int relu) {
    dim3 grid(N / 64, Msz / 128);
    k_gemm_bf16<<<grid, 256, 0, stream>>>(A, lda, Bt, K, bias, resid, ldr,
                                          Cf, ldc, Cb, ldcb, K, Nstore, relu);
  };

  for (int l = 0; l < Lsz; ++l) {
    // pre-LN 1
    k_ln_bf16<<<Msz / 4, 128, 0, stream>>>(x, ln1_g + l * Esz, ln1_b + l * Esz, bufA, Esz);
    // qkv projection: [M,384] x [384,1152] -> bf16 qkv
    gemm(bufA, Esz, wqkv_t + (size_t)l * QKVW * Esz, QKVW, Esz,
         nullptr, nullptr, 0, nullptr, 0, bufB, QKVW, QKVW, 0);
    // V transpose for the P@V WMMA
    k_vtrans<<<blocks((long)Bsz * Hsz * HSsz * Tsz), 256, 0, stream>>>(bufB, vTbuf);
    // attention -> bufA [M,384] bf16
    k_attn<<<Bsz * Hsz * (Tsz / 16), 32, 0, stream>>>(bufB, vTbuf, bufA);
    // output projection + residual into x
    gemm(bufA, Esz, wproj_t + (size_t)l * Esz * Esz, Esz, Esz,
         b_proj + l * Esz, x, Esz, x, Esz, nullptr, 0, Esz, 0);
    // pre-LN 2
    k_ln_bf16<<<Msz / 4, 128, 0, stream>>>(x, ln2_g + l * Esz, ln2_b + l * Esz, bufA, Esz);
    // ff1 + ReLU -> bf16 hidden
    gemm(bufA, Esz, w1_t + (size_t)l * FFsz * Esz, FFsz, Esz,
         b1 + l * FFsz, nullptr, 0, nullptr, 0, bufB, FFsz, FFsz, 1);
    // ff2 + residual into x
    gemm(bufB, FFsz, w2_t + (size_t)l * Esz * FFsz, Esz, FFsz,
         b2 + l * Esz, x, Esz, x, Esz, nullptr, 0, Esz, 0);
  }

  // final LN + head (N padded to 128, stores guarded to V=78)
  k_ln_bf16<<<Msz / 4, 128, 0, stream>>>(x, lnf_g, lnf_b, bufA, Esz);
  gemm(bufA, Esz, whead_t, 128, Esz,
       b_head, nullptr, 0, (float*)d_out, Vsz, nullptr, 0, Vsz, 0);
}
